// _CompatAttention_79164837200292
// MI455X (gfx1250) — matrix operand order repaired and checked by reading, not compiled, not run on hardware
//
#include <hip/hip_runtime.h>

// ---------------------------------------------------------------- types
typedef __bf16 bf16;
typedef __attribute__((ext_vector_type(16))) __bf16 v16bf;
typedef __attribute__((ext_vector_type(8)))  float  v8f;

#define Dm  2048
#define Hh  16
#define KVh 4
#define HDd 128
#define Bb  2
#define Ll  2048
#define BL  (Bb * Ll)
#define SCALE 0.08838834764831843f  // 1/sqrt(128)

// Load 16 bf16 (two 16B chunks) into a WMMA operand register set.
__device__ inline v16bf ld16(const bf16* p0, const bf16* p1) {
  union { v16bf v; uint4 u[2]; } t;
  t.u[0] = *reinterpret_cast<const uint4*>(p0);
  t.u[1] = *reinterpret_cast<const uint4*>(p1);
  return t.v;
}

__device__ inline v8f wmma_bf16(v16bf a, v16bf b, v8f c) {
  return __builtin_amdgcn_wmma_f32_16x16x32_bf16(
      false, a, false, b, (short)0, c, false, false);
}

// Async copy 16B global -> LDS (tracked by ASYNCcnt). The low 32 bits of a
// generic shared-memory pointer are the group-segment offset the HW wants.
__device__ inline void async_ld128(void* l, const void* g) {
  uint32_t lo = (uint32_t)(uintptr_t)l;
  asm volatile("global_load_async_to_lds_b128 %0, %1, off"
               :: "v"(lo), "v"(g) : "memory");
}
#define WAIT_ASYNC(n) asm volatile("s_wait_asynccnt " #n ::: "memory")

// ---------------------------------------------------------------- cvt f32 -> bf16
__global__ void cvt_kernel(const float* __restrict__ in, bf16* __restrict__ out, int n) {
  int i = blockIdx.x * blockDim.x + threadIdx.x;
  int stride = gridDim.x * blockDim.x;
  for (; i < n; i += stride) out[i] = (bf16)in[i];
}

// ---------------------------------------------------------------- GEMM C(M,N) = A(M,K) * B(N,K)^T
// 4 waves / block; block tile 64(M) x 128(N); K-step 32; double-buffered LDS
// filled with async global->LDS copies. Each wave computes a 32x64 subtile.
__global__ __launch_bounds__(128) void gemm_bf16_nt(
    const bf16* __restrict__ A, const bf16* __restrict__ B, float* __restrict__ C,
    int M, int N, int K) {
  __shared__ bf16 sA[2][64 * 32];    //  8 KB
  __shared__ bf16 sB[2][128 * 32];   // 16 KB

  int tid  = threadIdx.x;
  int lane = tid & 31, w = tid >> 5;
  int l15  = lane & 15;
  int offA = (lane & 16) ? 16 : 0;
  int offB = (lane & 16) ? 16 : 0;
  int rb   = (lane & 16) ? 8  : 0;
  int mh = (w & 1) * 32, nh = (w >> 1) * 64;
  int m0 = blockIdx.y * 64, n0 = blockIdx.x * 128;

  v8f acc[2][4] = {};

  auto stage = [&](int buf, int kc) {
#pragma unroll
    for (int r = 0; r < 2; ++r) {                 // A tile: 64 x 32
      int i = tid + 128 * r;
      int row = i >> 2, c = (i & 3) * 8;
      async_ld128(&sA[buf][row * 32 + c], A + (size_t)(m0 + row) * K + kc + c);
    }
#pragma unroll
    for (int r = 0; r < 4; ++r) {                 // B tile: 128 x 32
      int i = tid + 128 * r;
      int row = i >> 2, c = (i & 3) * 8;
      async_ld128(&sB[buf][row * 32 + c], B + (size_t)(n0 + row) * K + kc + c);
    }
  };

  int nk = K >> 5;
  stage(0, 0);
  for (int ks = 0; ks < nk; ++ks) {
    int cur = ks & 1;
    if (ks + 1 < nk) {
      stage(cur ^ 1, (ks + 1) << 5);   // prefetch next stage
      WAIT_ASYNC(6);                   // wait for current stage only
    } else {
      WAIT_ASYNC(0);
    }
    __syncthreads();                   // tile visible to all waves

    v16bf a[2], b[4];
#pragma unroll
    for (int mt = 0; mt < 2; ++mt) {
      const bf16* p = &sA[cur][(mh + mt * 16 + l15) * 32 + offA];
      a[mt] = ld16(p, p + 8);
    }
#pragma unroll
    for (int nt = 0; nt < 4; ++nt) {
      const bf16* p = &sB[cur][(nh + nt * 16 + l15) * 32 + offB];
      b[nt] = ld16(p, p + 8);
    }
#pragma unroll
    for (int mt = 0; mt < 2; ++mt)
#pragma unroll
      for (int nt = 0; nt < 4; ++nt)
        acc[mt][nt] = wmma_bf16(a[mt], b[nt], acc[mt][nt]);
    __syncthreads();                   // all waves done reading before re-stage
  }

#pragma unroll
  for (int mt = 0; mt < 2; ++mt)
#pragma unroll
    for (int nt = 0; nt < 4; ++nt) {
      float* cp = C + (size_t)(m0 + mh + mt * 16 + rb) * N + (n0 + nh + nt * 16 + l15);
#pragma unroll
      for (int v = 0; v < 8; ++v) cp[(size_t)v * N] = acc[mt][nt][v];
    }
}

// ---------------------------------------------------------------- fused RMSNorm + RoPE, f32 -> bf16
// One wave per (token, head). Layout: in/out (B*L, Hn, 128).
__global__ void normrope_kernel(const float* __restrict__ in, const float* __restrict__ w,
                                bf16* __restrict__ out, int total, int Hn) {
  int wid  = (blockIdx.x * blockDim.x + threadIdx.x) >> 5;
  int lane = threadIdx.x & 31;
  if (wid >= total) return;

  const float4 vv = reinterpret_cast<const float4*>(in + (size_t)wid * HDd)[lane];
  float ss = vv.x * vv.x + vv.y * vv.y + vv.z * vv.z + vv.w * vv.w;
#pragma unroll
  for (int m = 1; m < 32; m <<= 1) ss += __shfl_xor(ss, m, 32);
  float rstd = rsqrtf(ss * (1.0f / 128.0f) + 1e-6f);

  int pos = (wid / Hn) % Ll;
  int d0  = lane * 4;
  float x0 = vv.x * rstd * w[d0 + 0];
  float x1 = vv.y * rstd * w[d0 + 1];
  float x2 = vv.z * rstd * w[d0 + 2];
  float x3 = vv.w * rstd * w[d0 + 3];

  int   p0 = d0 >> 1;
  float f0 = __powf(10000.0f, -(float)p0       * (1.0f / 64.0f));
  float f1 = __powf(10000.0f, -(float)(p0 + 1) * (1.0f / 64.0f));
  float a0 = (float)pos * f0, a1 = (float)pos * f1;
  float c0 = __cosf(a0), s0 = __sinf(a0);
  float c1 = __cosf(a1), s1 = __sinf(a1);

  bf16* op = out + (size_t)wid * HDd + d0;
  op[0] = (bf16)(x0 * c0 - x1 * s0);
  op[1] = (bf16)(x0 * s0 + x1 * c0);
  op[2] = (bf16)(x2 * c1 - x3 * s1);
  op[3] = (bf16)(x2 * s1 + x3 * c1);
}

// ---------------------------------------------------------------- V transpose: (B*L,KV,HD) f32 -> (B,KV,HD,L) bf16
__global__ void vtrans_kernel(const float* __restrict__ v, bf16* __restrict__ vt, int n) {
  int i = blockIdx.x * blockDim.x + threadIdx.x;
  if (i >= n) return;
  int    hd  = i & (HDd - 1);
  int    kvh = (i >> 7) & (KVh - 1);
  int    tok = i >> 9;
  int    b   = tok / Ll;
  int    l   = tok % Ll;
  vt[(((size_t)b * KVh + kvh) * HDd + hd) * Ll + l] = (bf16)v[i];
}

// ---------------------------------------------------------------- flash attention (causal, GQA)
// grid (L/64, H, B), block 128 (4 waves). Wave w owns query rows [qt*64+w*16, +16).
// K (64x128) and V (128x64, hd-major) tiles are double-buffered in LDS via
// async copies and shared by all 4 waves.
__global__ __launch_bounds__(128) void attn_kernel(
    const bf16* __restrict__ qb,   // (B*L, H, HD)
    const bf16* __restrict__ kb,   // (B*L, KV, HD)
    const bf16* __restrict__ vt,   // (B, KV, HD, L)
    float* __restrict__ o) {       // (B*L, H, HD)
  __shared__ bf16 sK[2][64 * 128];   // 32 KB
  __shared__ bf16 sV[2][128 * 64];   // 32 KB
  __shared__ bf16 sP[4][16 * 64];    //  8 KB

  int qt = blockIdx.x, h = blockIdx.y, b = blockIdx.z;
  int tid = threadIdx.x;
  int w = tid >> 5, lane = tid & 31;
  int l15 = lane & 15;
  int offA = (lane & 16) ? 16 : 0;
  int offB = (lane & 16) ? 16 : 0;
  int rb   = (lane & 16) ? 8 : 0;
  int kv = h >> 2;
  int qrow0 = qt * 64 + w * 16;

  const bf16* kbase = kb + ((size_t)b * Ll * KVh + kv) * HDd;
  const bf16* vbase = vt + ((size_t)b * KVh + kv) * HDd * (size_t)Ll;

  auto stage_kv = [&](int buf, int jt) {
#pragma unroll
    for (int r = 0; r < 8; ++r) {                  // K tile: 64 keys x 128 hd
      int i = tid + 128 * r;
      int row = i >> 4, c = (i & 15) * 8;
      async_ld128(&sK[buf][row * 128 + c],
                  kbase + (size_t)(jt * 64 + row) * (KVh * HDd) + c);
    }
#pragma unroll
    for (int r = 0; r < 8; ++r) {                  // V tile: 128 hd x 64 keys
      int i = tid + 128 * r;
      int row = i >> 3, c = (i & 7) * 8;
      async_ld128(&sV[buf][row * 64 + c],
                  vbase + (size_t)row * Ll + jt * 64 + c);
    }
  };

  // Q operand (reused across all key tiles): 16 rows x HD=128 (4 K-chunks).
  const bf16* qrowp = qb + ((size_t)(b * Ll + qrow0 + l15) * Hh + h) * HDd;
  v16bf qA[4];
#pragma unroll
  for (int kc = 0; kc < 4; ++kc) {
    const bf16* p = qrowp + kc * 32 + offA;
    qA[kc] = ld16(p, p + 8);
  }

  float Mv[8], Sv[8];
#pragma unroll
  for (int v = 0; v < 8; ++v) { Mv[v] = -3.0e38f; Sv[v] = 0.0f; }
  v8f oA[8] = {};

  bf16* lp = &sP[w][0];

  stage_kv(0, 0);
  for (int jt = 0; jt <= qt; ++jt) {
    int cur = jt & 1;
    if (jt < qt) {
      stage_kv(cur ^ 1, jt + 1);   // prefetch next K/V tile
      WAIT_ASYNC(16);              // wait for current tile only
    } else {
      WAIT_ASYNC(0);
    }
    __syncthreads();               // tile visible to all waves

    // ---- S strip = Q * K^T  (16 x 64)
    v8f s[4] = {};
#pragma unroll
    for (int nt = 0; nt < 4; ++nt)
#pragma unroll
      for (int kc = 0; kc < 4; ++kc) {
        const bf16* p = &sK[cur][(nt * 16 + l15) * 128 + kc * 32 + offB];
        v16bf bk = ld16(p, p + 8);
        s[nt] = wmma_bf16(qA[kc], bk, s[nt]);
      }
#pragma unroll
    for (int nt = 0; nt < 4; ++nt) s[nt] *= SCALE;

    if (jt == qt) {  // causal mask inside the diagonal tile
#pragma unroll
      for (int nt = 0; nt < 4; ++nt)
#pragma unroll
        for (int v = 0; v < 8; ++v)
          if (jt * 64 + nt * 16 + l15 > qrow0 + rb + v) s[nt][v] = -3.0e38f;
    }

    // ---- online softmax
    float mnew[8], alpha[8], rs[8];
#pragma unroll
    for (int v = 0; v < 8; ++v) {
      float t = fmaxf(fmaxf(s[0][v], s[1][v]), fmaxf(s[2][v], s[3][v]));
#pragma unroll
      for (int m = 1; m < 16; m <<= 1) t = fmaxf(t, __shfl_xor(t, m, 32));
      mnew[v]  = fmaxf(Mv[v], t);
      alpha[v] = __expf(Mv[v] - mnew[v]);
      Mv[v]    = mnew[v];
      rs[v]    = 0.0f;
    }
#pragma unroll
    for (int nt = 0; nt < 4; ++nt)
#pragma unroll
      for (int v = 0; v < 8; ++v) {
        float p = __expf(s[nt][v] - mnew[v]);
        s[nt][v] = p;
        rs[v] += p;
      }
#pragma unroll
    for (int v = 0; v < 8; ++v) {
      float t = rs[v];
#pragma unroll
      for (int m = 1; m < 16; m <<= 1) t += __shfl_xor(t, m, 32);
      Sv[v] = Sv[v] * alpha[v] + t;
    }
#pragma unroll
    for (int ht = 0; ht < 8; ++ht)
#pragma unroll
      for (int v = 0; v < 8; ++v) oA[ht][v] *= alpha[v];

    // ---- transpose P (D-layout -> A-layout) through per-wave LDS tile.
    // Same-wave DS ops complete in order; no barrier needed here.
#pragma unroll
    for (int nt = 0; nt < 4; ++nt)
#pragma unroll
      for (int v = 0; v < 8; ++v)
        lp[(rb + v) * 64 + nt * 16 + l15] = (bf16)s[nt][v];

    v16bf pA[2];
#pragma unroll
    for (int kc2 = 0; kc2 < 2; ++kc2) {
      const bf16* p = lp + l15 * 64 + kc2 * 32 + offA;
      pA[kc2] = ld16(p, p + 8);
    }

    // ---- O += P * V  (V tile in LDS, hd-major: per-lane contiguous keys)
#pragma unroll
    for (int kc2 = 0; kc2 < 2; ++kc2)
#pragma unroll
      for (int ht = 0; ht < 8; ++ht) {
        const bf16* p = &sV[cur][(ht * 16 + l15) * 64 + kc2 * 32 + offB];
        v16bf vb = ld16(p, p + 8);
        oA[ht] = wmma_bf16(pA[kc2], vb, oA[ht]);
      }
    __syncthreads();  // all waves done with this K/V buffer before re-stage
  }

  // ---- epilogue: normalize and store
  float inv[8];
#pragma unroll
  for (int v = 0; v < 8; ++v) inv[v] = 1.0f / Sv[v];
#pragma unroll
  for (int ht = 0; ht < 8; ++ht)
#pragma unroll
    for (int v = 0; v < 8; ++v) {
      int qr = qrow0 + rb + v;
      o[((size_t)(b * Ll + qr) * Hh + h) * HDd + ht * 16 + l15] = oA[ht][v] * inv[v];
    }
}

// ---------------------------------------------------------------- launch
extern "C" void kernel_launch(void* const* d_in, const int* in_sizes, int n_in,
                              void* d_out, int out_size, void* d_ws, size_t ws_size,
                              hipStream_t stream) {
  const float* x   = (const float*)d_in[0];
  const float* wq  = (const float*)d_in[1];
  const float* wk  = (const float*)d_in[2];
  const float* wv  = (const float*)d_in[3];
  const float* wo  = (const float*)d_in[4];
  const float* qnw = (const float*)d_in[5];
  const float* knw = (const float*)d_in[6];
  float* out = (float*)d_out;

  char* ws = (char*)d_ws;
  bf16* xb  = (bf16*)(ws + 0);                              // 16.78 MB
  bf16* wqb = (bf16*)(ws + 16777216);                       //  8.39 MB
  bf16* wkb = (bf16*)(ws + 25165824);                       //  2.10 MB
  bf16* wvb = (bf16*)(ws + 27262976);                       //  2.10 MB
  bf16* wob = (bf16*)(ws + 29360128);                       //  8.39 MB
  float* qf = (float*)(ws + 37748736);                      // 33.55 MB
  float* kf = (float*)(ws + 71303168);                      //  8.39 MB
  float* vf = (float*)(ws + 79691776);                      //  8.39 MB
  bf16* qbn = (bf16*)(ws + 88080384);                       // 16.78 MB
  bf16* kbn = (bf16*)(ws + 104857600);                      //  4.19 MB
  bf16* vtb = (bf16*)(ws + 109051904);                      //  4.19 MB
  float* of = qf;   // alias: q_f32 dead after normrope
  bf16*  ob = xb;   // alias: x_bf16 dead after QKV gemms

  const int TB = 256;
  auto blocks = [](int n, int t) { return (n + t - 1) / t; };

  // 1) convert inputs to bf16
  cvt_kernel<<<blocks(BL * Dm, TB), TB, 0, stream>>>(x,  xb,  BL * Dm);
  cvt_kernel<<<blocks(Dm * Dm, TB), TB, 0, stream>>>(wq, wqb, Dm * Dm);
  cvt_kernel<<<blocks(512 * Dm, TB), TB, 0, stream>>>(wk, wkb, 512 * Dm);
  cvt_kernel<<<blocks(512 * Dm, TB), TB, 0, stream>>>(wv, wvb, 512 * Dm);
  cvt_kernel<<<blocks(Dm * Dm, TB), TB, 0, stream>>>(wo, wob, Dm * Dm);

  // 2) QKV projections (WMMA GEMM, async-LDS staged)
  gemm_bf16_nt<<<dim3(Dm / 128,  BL / 64), 128, 0, stream>>>(xb, wqb, qf, BL, Dm,  Dm);
  gemm_bf16_nt<<<dim3(512 / 128, BL / 64), 128, 0, stream>>>(xb, wkb, kf, BL, 512, Dm);
  gemm_bf16_nt<<<dim3(512 / 128, BL / 64), 128, 0, stream>>>(xb, wvb, vf, BL, 512, Dm);

  // 3) RMSNorm + RoPE (q, k) -> bf16 ; V -> transposed bf16
  normrope_kernel<<<blocks(BL * Hh  * 32, TB), TB, 0, stream>>>(qf, qnw, qbn, BL * Hh,  Hh);
  normrope_kernel<<<blocks(BL * KVh * 32, TB), TB, 0, stream>>>(kf, knw, kbn, BL * KVh, KVh);
  vtrans_kernel<<<blocks(BL * 512, TB), TB, 0, stream>>>(vf, vtb, BL * 512);

  // 4) causal flash attention (async-LDS staged K/V)
  attn_kernel<<<dim3(Ll / 64, Hh, Bb), 128, 0, stream>>>(qbn, kbn, vtb, of);

  // 5) output projection
  cvt_kernel<<<blocks(BL * Dm, TB), TB, 0, stream>>>(of, ob, BL * Dm);
  gemm_bf16_nt<<<dim3(Dm / 128, BL / 64), 128, 0, stream>>>(ob, wob, out, BL, Dm, Dm);
}
